// GNNLayer_249108103824
// MI455X (gfx1250) — matrix operand order repaired and checked by reading, not compiled, not run on hardware
//
#include <hip/hip_runtime.h>
#include <hip/hip_bf16.h>
#include <math.h>

// ---------------------------------------------------------------------------
// Single-query attention + FFN layer for MI455X (gfx1250), wave32, WMMA.
//
// D=1024, H=16, dh=64, K=32768.
// Algebraic refactor: scores = N @ P^T  (P[h] = Wk_h^T q_h / 8, 16x1024),
//                     u      = attn @ N (16x1024), ctx_h = Wv_h u_h + bv_h.
// => only two passes over neighbors (134 MB each; 2nd pass L2-resident),
//    both done with v_wmma_f32_16x16x32_bf16, fp32 accumulation.
// Second pass is parallelized 16 column-slabs x 16 K-chunks = 256 WGs with a
// deterministic partial-sum reduction (no float atomics).
// ---------------------------------------------------------------------------

typedef __attribute__((ext_vector_type(16))) __bf16          v16bf;
typedef __attribute__((ext_vector_type(16))) unsigned short  v16u;
typedef __attribute__((ext_vector_type(8)))  unsigned short  v8u;
typedef __attribute__((ext_vector_type(8)))  float           v8f;

#define D_DIM 1024
#define H_DIM 16
#define K_DIM 32768
#define KCHUNKS 16

__device__ __forceinline__ unsigned short f2bf(float f) {
    unsigned int u = __builtin_bit_cast(unsigned int, f);
    unsigned int r = u + 0x7FFFu + ((u >> 16) & 1u);   // round-to-nearest-even
    return (unsigned short)(r >> 16);
}

__device__ __forceinline__ float wave_reduce_add(float v) {
    #pragma unroll
    for (int o = 16; o > 0; o >>= 1) v += __shfl_down(v, o, 32);
    return v;
}

// ---------------------------------------------------------------------------
// Generic row-per-wave matvec: out[r] = act(dot(W[r,:], x) + bias[r])
// cols must be a multiple of 128. act: 0 = none, 1 = exact GELU.
// ---------------------------------------------------------------------------
__global__ __launch_bounds__(256)
void matvec_kernel(const float* __restrict__ W, const float* __restrict__ x,
                   const float* __restrict__ bias, float* __restrict__ out,
                   int rows, int cols, int act) {
    int wid  = threadIdx.x >> 5;
    int lane = threadIdx.x & 31;
    int r = blockIdx.x * 8 + wid;
    if (r >= rows) return;
    const float* Wr = W + (size_t)r * cols;
    float acc = 0.f;
    for (int c = lane * 4; c < cols; c += 128) {
        float4 w4 = *(const float4*)(Wr + c);
        float4 x4 = *(const float4*)(x + c);
        acc = fmaf(w4.x, x4.x, acc);
        acc = fmaf(w4.y, x4.y, acc);
        acc = fmaf(w4.z, x4.z, acc);
        acc = fmaf(w4.w, x4.w, acc);
    }
    acc = wave_reduce_add(acc);
    if (lane == 0) {
        float v = acc + (bias ? bias[r] : 0.f);
        if (act == 1) v = 0.5f * v * (1.f + erff(v * 0.70710678118654752f));
        out[r] = v;
    }
}

// ---------------------------------------------------------------------------
// P[h][c] = (1/8) * sum_d q[h*64+d] * Wk[h*64+d][c], stored as bf16 row-major.
// grid = (D/256, H), block = 256. Coalesced along c.
// ---------------------------------------------------------------------------
__global__ __launch_bounds__(256)
void build_p_kernel(const float* __restrict__ Wk, const float* __restrict__ q,
                    unsigned short* __restrict__ Pbf) {
    int h = blockIdx.y;
    int c = blockIdx.x * 256 + threadIdx.x;
    const float* base = Wk + (size_t)(h * 64) * D_DIM + c;
    float acc = 0.f;
    #pragma unroll 8
    for (int d = 0; d < 64; ++d)
        acc = fmaf(q[h * 64 + d], base[(size_t)d * D_DIM], acc);
    Pbf[h * D_DIM + c] = f2bf(acc * 0.125f);
}

// ---------------------------------------------------------------------------
// scores[h][k] via WMMA: one 16-row x 16-head tile per wave, reduce over 1024.
// A = neighbors rows (fp32 -> bf16 in-register), B = P^T (bf16 preconverted).
// grid = K/16/8 blocks of 256 threads (8 waves).
// ---------------------------------------------------------------------------
__global__ __launch_bounds__(256)
void scores_kernel(const float* __restrict__ nb, const unsigned short* __restrict__ Pbf,
                   float* __restrict__ sbuf) {
    int wid  = threadIdx.x >> 5;
    int lane = threadIdx.x & 31;
    int tile = blockIdx.x * 8 + wid;
    int lm   = lane & 15;       // A: neighbor row in tile;  B: head column
    int hi   = lane >> 4;       // half-wave select
    const float*          arow = nb + ((size_t)tile * 16 + lm) * D_DIM;
    const unsigned short* brow = Pbf + (size_t)lm * D_DIM;
    int aoff = hi ? 16 : 0;     // A K-elements: 0..15 | 16..31
    int boff = hi ? 16 : 0;     // B K-elements: 0..15 | 16..31

    v8f acc = {0.f, 0.f, 0.f, 0.f, 0.f, 0.f, 0.f, 0.f};
    for (int c0 = 0; c0 < D_DIM; c0 += 32) {
        if (c0 + 128 < D_DIM) __builtin_prefetch(arow + c0 + 128, 0, 1);
        float4 a0 = *(const float4*)(arow + c0 + aoff);
        float4 a1 = *(const float4*)(arow + c0 + aoff + 4);
        float4 a2 = *(const float4*)(arow + c0 + aoff + 8);
        float4 a3 = *(const float4*)(arow + c0 + aoff + 12);
        union { v16u v; unsigned short e[16]; } ua;
        ua.e[0]  = f2bf(a0.x); ua.e[1]  = f2bf(a0.y); ua.e[2]  = f2bf(a0.z); ua.e[3]  = f2bf(a0.w);
        ua.e[4]  = f2bf(a1.x); ua.e[5]  = f2bf(a1.y); ua.e[6]  = f2bf(a1.z); ua.e[7]  = f2bf(a1.w);
        ua.e[8]  = f2bf(a2.x); ua.e[9]  = f2bf(a2.y); ua.e[10] = f2bf(a2.z); ua.e[11] = f2bf(a2.w);
        ua.e[12] = f2bf(a3.x); ua.e[13] = f2bf(a3.y); ua.e[14] = f2bf(a3.z); ua.e[15] = f2bf(a3.w);
        v16bf av = __builtin_bit_cast(v16bf, ua.v);
        v16u  bu = *(const v16u*)(brow + c0 + boff);
        v16bf bv = __builtin_bit_cast(v16bf, bu);
        acc = __builtin_amdgcn_wmma_f32_16x16x32_bf16(false, av, false, bv,
                                                      (short)0, acc, false, false);
    }
    // D tile: M = v + 8*hi (neighbor row), N = lane&15 (head)
    #pragma unroll
    for (int v = 0; v < 8; ++v)
        sbuf[(size_t)lm * K_DIM + tile * 16 + v + hi * 8] = acc[v];
}

// ---------------------------------------------------------------------------
// Per-head softmax over K: computes max, sum(exp), writes normalized bf16
// weights. One block (256 threads) per head; deterministic tree reductions.
// ---------------------------------------------------------------------------
__global__ __launch_bounds__(256)
void softmax_kernel(const float* __restrict__ sbuf, unsigned short* __restrict__ wbuf) {
    __shared__ float red[256];
    __shared__ float s_m, s_inv;
    int h = blockIdx.x;
    const float* s = sbuf + (size_t)h * K_DIM;
    float mx = -INFINITY;
    for (int k = threadIdx.x; k < K_DIM; k += 256) mx = fmaxf(mx, s[k]);
    red[threadIdx.x] = mx; __syncthreads();
    for (int o = 128; o > 0; o >>= 1) {
        if (threadIdx.x < o) red[threadIdx.x] = fmaxf(red[threadIdx.x], red[threadIdx.x + o]);
        __syncthreads();
    }
    if (threadIdx.x == 0) s_m = red[0];
    __syncthreads();
    float m = s_m, sum = 0.f;
    for (int k = threadIdx.x; k < K_DIM; k += 256) sum += expf(s[k] - m);
    red[threadIdx.x] = sum; __syncthreads();
    for (int o = 128; o > 0; o >>= 1) {
        if (threadIdx.x < o) red[threadIdx.x] += red[threadIdx.x + o];
        __syncthreads();
    }
    if (threadIdx.x == 0) s_inv = 1.f / red[0];
    __syncthreads();
    float inv = s_inv;
    for (int k = threadIdx.x; k < K_DIM; k += 256)
        wbuf[(size_t)h * K_DIM + k] = f2bf(expf(s[k] - m) * inv);
}

// ---------------------------------------------------------------------------
// upart[kc] (16 x 1024) = weights (16 x Kchunk, bf16) @ neighbors chunk via
// WMMA. grid = (16 column slabs, 16 K-chunks) = 256 WGs for full-device
// bandwidth. 8 waves split the K-chunk; LDS tree-combine within the block;
// partials summed by reduce_u_kernel (deterministic).
// ---------------------------------------------------------------------------
__global__ __launch_bounds__(256)
void ctxsum_kernel(const unsigned short* __restrict__ wbuf, const float* __restrict__ nb,
                   float* __restrict__ upart) {
    __shared__ float part[8][4][16][16];   // 32 KB
    int wid  = threadIdx.x >> 5;
    int lane = threadIdx.x & 31;
    int hi   = lane >> 4;
    int lm   = lane & 15;       // A: head row; B/D: column index n
    int colbase = blockIdx.x * 64;
    int aoff = hi ? 16 : 0;
    int kk0  = hi ? 16 : 0;
    const int kspan = K_DIM / KCHUNKS;            // 2048 rows per block
    int kbeg = blockIdx.y * kspan + wid * (kspan / 8);
    int kend = kbeg + (kspan / 8);                // 256 rows per wave

    v8f zero = {0.f, 0.f, 0.f, 0.f, 0.f, 0.f, 0.f, 0.f};
    v8f acc[4] = {zero, zero, zero, zero};

    const unsigned short* wrow = wbuf + (size_t)lm * K_DIM;
    for (int k0 = kbeg; k0 < kend; k0 += 32) {
        union { v16u v; v8u h2[2]; } ua;
        ua.h2[0] = *(const v8u*)(wrow + k0 + aoff);
        ua.h2[1] = *(const v8u*)(wrow + k0 + aoff + 8);
        v16bf av = __builtin_bit_cast(v16bf, ua.v);
        #pragma unroll
        for (int t = 0; t < 4; ++t) {
            int ct = colbase + t * 16;
            union { v16u v; unsigned short e[16]; } ub;
            #pragma unroll
            for (int j = 0; j < 16; ++j)
                ub.e[j] = f2bf(nb[(size_t)(k0 + kk0 + j) * D_DIM + ct + lm]);
            v16bf bv = __builtin_bit_cast(v16bf, ub.v);
            acc[t] = __builtin_amdgcn_wmma_f32_16x16x32_bf16(false, av, false, bv,
                                                             (short)0, acc[t], false, false);
        }
    }
    #pragma unroll
    for (int t = 0; t < 4; ++t)
        #pragma unroll
        for (int v = 0; v < 8; ++v)
            part[wid][t][v + hi * 8][lm] = acc[t][v];
    __syncthreads();
    int n = threadIdx.x & 15;
    int m = threadIdx.x >> 4;      // head
    float* up = upart + (size_t)blockIdx.y * (H_DIM * D_DIM);
    #pragma unroll
    for (int t = 0; t < 4; ++t) {
        float s = 0.f;
        #pragma unroll
        for (int w = 0; w < 8; ++w) s += part[w][t][m][n];
        up[(size_t)m * D_DIM + colbase + t * 16 + n] = s;
    }
}

// ---------------------------------------------------------------------------
// u[idx] = sum over 16 K-chunk partials. idx in [0, 16*1024).
// ---------------------------------------------------------------------------
__global__ __launch_bounds__(256)
void reduce_u_kernel(const float* __restrict__ upart, float* __restrict__ u) {
    int idx = blockIdx.x * 256 + threadIdx.x;
    float s = 0.f;
    #pragma unroll
    for (int p = 0; p < KCHUNKS; ++p)
        s += upart[(size_t)p * (H_DIM * D_DIM) + idx];
    u[idx] = s;
}

// ---------------------------------------------------------------------------
// ctx[r] = dot(Wv[r,:], u[r/64,:]) + bv[r]   (per-head segment of u)
// ---------------------------------------------------------------------------
__global__ __launch_bounds__(256)
void ctxproj_kernel(const float* __restrict__ Wv, const float* __restrict__ bv,
                    const float* __restrict__ u, float* __restrict__ ctx) {
    int wid  = threadIdx.x >> 5;
    int lane = threadIdx.x & 31;
    int r = blockIdx.x * 8 + wid;
    const float* Wr = Wv + (size_t)r * D_DIM;
    const float* ur = u + (size_t)(r >> 6) * D_DIM;
    float acc = 0.f;
    for (int c = lane * 4; c < D_DIM; c += 128) {
        float4 w4 = *(const float4*)(Wr + c);
        float4 u4 = *(const float4*)(ur + c);
        acc = fmaf(w4.x, u4.x, acc);
        acc = fmaf(w4.y, u4.y, acc);
        acc = fmaf(w4.z, u4.z, acc);
        acc = fmaf(w4.w, u4.w, acc);
    }
    acc = wave_reduce_add(acc);
    if (lane == 0) ctx[r] = acc + bv[r];
}

// ---------------------------------------------------------------------------
// out = LayerNorm(a + b) * g + be   (D=1024; 1 block, 256 threads x 4 elems)
// ---------------------------------------------------------------------------
__global__ __launch_bounds__(256)
void ln_kernel(const float* __restrict__ a, const float* __restrict__ b,
               const float* __restrict__ g, const float* __restrict__ be,
               float* __restrict__ out) {
    __shared__ float s1[256], s2[256];
    float v[4]; float sum = 0.f, sq = 0.f;
    #pragma unroll
    for (int i = 0; i < 4; ++i) {
        int idx = threadIdx.x + i * 256;
        float t = a[idx] + b[idx];
        v[i] = t; sum += t; sq += t * t;
    }
    s1[threadIdx.x] = sum; s2[threadIdx.x] = sq; __syncthreads();
    for (int o = 128; o > 0; o >>= 1) {
        if (threadIdx.x < o) { s1[threadIdx.x] += s1[threadIdx.x + o];
                               s2[threadIdx.x] += s2[threadIdx.x + o]; }
        __syncthreads();
    }
    float mean = s1[0] * (1.f / D_DIM);
    float var  = s2[0] * (1.f / D_DIM) - mean * mean;
    float rs   = rsqrtf(var + 1e-5f);
    #pragma unroll
    for (int i = 0; i < 4; ++i) {
        int idx = threadIdx.x + i * 256;
        out[idx] = (v[i] - mean) * rs * g[idx] + be[idx];
    }
}

// ---------------------------------------------------------------------------
extern "C" void kernel_launch(void* const* d_in, const int* in_sizes, int n_in,
                              void* d_out, int out_size, void* d_ws, size_t ws_size,
                              hipStream_t stream) {
    const float* query = (const float*)d_in[0];
    const float* nb    = (const float*)d_in[1];
    const float* ipw   = (const float*)d_in[2];   // (3D, D)
    const float* ipb   = (const float*)d_in[3];   // (3D,)
    const float* opw   = (const float*)d_in[4];
    const float* opb   = (const float*)d_in[5];
    const float* w1    = (const float*)d_in[6];
    const float* b1    = (const float*)d_in[7];
    const float* w2    = (const float*)d_in[8];
    const float* b2    = (const float*)d_in[9];
    const float* g1    = (const float*)d_in[10];
    const float* be1   = (const float*)d_in[11];
    const float* g2    = (const float*)d_in[12];
    const float* be2   = (const float*)d_in[13];
    float* out = (float*)d_out;

    char* ws = (char*)d_ws;
    float*          q     = (float*)(ws + 0x0);        //  4 KB
    unsigned short* Pbf   = (unsigned short*)(ws + 0x1000);   // 32 KB bf16
    float*          ctxb  = (float*)(ws + 0x9000);     //  4 KB
    float*          ao    = (float*)(ws + 0xA000);     //  4 KB
    float*          x     = (float*)(ws + 0xB000);     //  4 KB
    float*          y2    = (float*)(ws + 0xC000);     //  4 KB
    float*          h1    = (float*)(ws + 0xD000);     // 16 KB
    float*          u     = (float*)(ws + 0x11000);    // 64 KB
    float*          sbuf  = (float*)(ws + 0x21000);    //  2 MB
    unsigned short* wbuf  = (unsigned short*)(ws + 0x221000); // 1 MB
    float*          upart = (float*)(ws + 0x321000);   //  1 MB (16 partials)

    // 1) q = Wq @ query + bq
    matvec_kernel<<<128, 256, 0, stream>>>(ipw, query, ipb, q, D_DIM, D_DIM, 0);
    // 2) P = per-head (Wk_h^T q_h)/8, bf16
    build_p_kernel<<<dim3(4, 16), 256, 0, stream>>>(ipw + (size_t)D_DIM * D_DIM, q, Pbf);
    // 3) scores = neighbors @ P^T   (WMMA, pass 1 over neighbors)
    scores_kernel<<<K_DIM / 16 / 8, 256, 0, stream>>>(nb, Pbf, sbuf);
    // 4) per-head softmax -> bf16 weights
    softmax_kernel<<<H_DIM, 256, 0, stream>>>(sbuf, wbuf);
    // 5) u = weights @ neighbors    (WMMA, pass 2, 256 WGs, L2-resident)
    ctxsum_kernel<<<dim3(D_DIM / 64, KCHUNKS), 256, 0, stream>>>(wbuf, nb, upart);
    //    deterministic partial reduction
    reduce_u_kernel<<<(H_DIM * D_DIM) / 256, 256, 0, stream>>>(upart, u);
    // 6) ctx = Wv_h u_h + bv
    ctxproj_kernel<<<128, 256, 0, stream>>>(ipw + 2 * (size_t)D_DIM * D_DIM,
                                            ipb + 2 * D_DIM, u, ctxb);
    // 7) attn_out = Wout @ ctx + bout
    matvec_kernel<<<128, 256, 0, stream>>>(opw, ctxb, opb, ao, D_DIM, D_DIM, 0);
    // 8) x = LN(query + attn_out)
    ln_kernel<<<1, 256, 0, stream>>>(query, ao, g1, be1, x);
    // 9) h1 = gelu(W1 @ x + b1)
    matvec_kernel<<<512, 256, 0, stream>>>(w1, x, b1, h1, 4 * D_DIM, D_DIM, 1);
    // 10) y2 = W2 @ h1 + b2
    matvec_kernel<<<128, 256, 0, stream>>>(w2, h1, b2, y2, D_DIM, 4 * D_DIM, 0);
    // 11) out = LN(x + y2)
    ln_kernel<<<1, 256, 0, stream>>>(x, y2, g2, be2, out);
}
